// LSTM_2121713845065
// MI455X (gfx1250) — matrix operand order repaired and checked by reading, not compiled, not run on hardware
//
#include <hip/hip_runtime.h>
#include <hip/hip_bf16.h>

// ---------------------------------------------------------------------------
// 2-layer LayerNorm-LSTM + tied-embedding logits for MI455X (gfx1250, wave32)
// Matmuls run on v_wmma_f32_16x16x32_bf16 with fp32 accumulation.
// Recurrence: one fused kernel per (step, layer): in-block LN -> LDS A panel
// -> WMMA GEMM (K split 4-ways per gate) -> LDS reduce -> LSTM cell update.
// ---------------------------------------------------------------------------

#define BB 16
#define SS 512
#define NL 2
#define HH 1024
#define H2 2048
#define H4 4096
#define VV 32000

typedef __attribute__((ext_vector_type(16))) __bf16 v16bf;
typedef __attribute__((ext_vector_type(8)))  float  v8f;

union FragBF { uint4 u[2]; v16bf v; };

__device__ __forceinline__ unsigned short f32_to_bf16(float f) {
  unsigned int u = __float_as_uint(f);
  unsigned int r = u + 0x7fffu + ((u >> 16) & 1u);
  return (unsigned short)(r >> 16);
}

// A-matrix (16x32 bf16) fragment: row = lane%16 striped across K per ISA layout.
// lanes 0-15: K {0..7, 16..23}; lanes 16-31: K {8..15, 24..31}.
__device__ __forceinline__ v16bf load_a_frag(const unsigned short* row, int lane) {
  int klo = (lane & 16) ? 8 : 0;
  FragBF f;
  f.u[0] = *(const uint4*)(row + klo);
  f.u[1] = *(const uint4*)(row + klo + 16);
  return f.v;
}

// B-matrix (32x16 bf16) fragment from N-major storage (col = lane%16, K contiguous).
// lanes 0-15 hold K {0..7, 16..23}, lanes 16-31 hold K {8..15, 24..31}.
__device__ __forceinline__ v16bf load_b_frag(const unsigned short* col, int lane) {
  int klo = (lane & 16) ? 8 : 0;
  FragBF f;
  f.u[0] = *(const uint4*)(col + klo);
  f.u[1] = *(const uint4*)(col + klo + 16);
  return f.v;
}

// ---- prep: W (L,2H,4H) fp32 -> Wt (L,4H,2H) bf16 (N-major so B frags are linear)
__global__ void k_cvt_w(const float* __restrict__ W, unsigned short* __restrict__ Wt) {
  long long i = (long long)blockIdx.x * blockDim.x + threadIdx.x;
  if (i >= (long long)NL * H4 * H2) return;
  int k = (int)(i % H2);
  long long t = i / H2;
  int n = (int)(t % H4);
  int m = (int)(t / H4);
  Wt[i] = f32_to_bf16(W[((long long)m * H2 + k) * H4 + n]);
}

// ---- prep: embed fp32 -> bf16 (already (V,H) row-major = N-major for logits B)
__global__ void k_cvt_embed(const float* __restrict__ E, unsigned short* __restrict__ Eb) {
  long long i = (long long)blockIdx.x * blockDim.x + threadIdx.x;
  if (i >= (long long)VV * HH) return;
  Eb[i] = f32_to_bf16(E[i]);
}

// ---- prep: xe[t][b][h] = embed[x[b][t]][h]  (time-major fp32)
__global__ void k_gather(const int* __restrict__ x, const float* __restrict__ E,
                         float* __restrict__ xe) {
  long long i = (long long)blockIdx.x * blockDim.x + threadIdx.x;
  if (i >= (long long)SS * BB * HH) return;
  int h = (int)(i % HH);
  long long t2 = i / HH;
  int b = (int)(t2 % BB);
  int t = (int)(t2 / BB);
  int tok = x[b * SS + t];
  xe[i] = E[(long long)tok * HH + h];
}

// ---- fused per-(step,layer) kernel. Grid = 64 blocks (16-wide hidden tiles),
// 512 threads = 16 waves.
//  Phase 1: wave-per-row LayerNorm (shfl reductions) + h_prev -> bf16 A[16][2048] in LDS
//  Phase 2: wave = (gate g, K-quarter kq); 16 WMMA k-steps, B streamed from L2
//  Phase 3: LDS reused for partial-C reduce; 256 threads do the LSTM cell update.
__global__ void k_lstm_step(const float* __restrict__ inp,
                            const float* __restrict__ gamma,
                            const float* __restrict__ beta,
                            const float* __restrict__ hprev,
                            const unsigned short* __restrict__ Wm,
                            const float* __restrict__ bias,
                            float* __restrict__ cstate,
                            float* __restrict__ hout,
                            const float* __restrict__ resid,   // layer0: xe[t]
                            float* __restrict__ inp_out,       // layer0: inp_cur
                            unsigned short* __restrict__ hseq_out) { // layer1: hseq + t*H
  __shared__ __align__(16) unsigned short smem[16 * H2];   // 64 KB: A panel, then C
  int tid = threadIdx.x;
  int lane = tid & 31;
  int w = tid >> 5;                      // 0..15

  // ---- Phase 1: LN + concat into LDS (wave w owns row w)
  {
    const float* row = inp + w * HH;
    float sum = 0.f, sq = 0.f;
    for (int e = lane; e < HH; e += 32) { float v = row[e]; sum += v; sq += v * v; }
    for (int m = 16; m > 0; m >>= 1) {
      sum += __shfl_xor(sum, m);
      sq  += __shfl_xor(sq, m);
    }
    float mu = sum * (1.f / HH);
    float var = sq * (1.f / HH) - mu * mu;
    float rs = rsqrtf(var + 1e-6f);
    for (int e = lane; e < HH; e += 32)
      smem[w * H2 + e] = f32_to_bf16((row[e] - mu) * rs * gamma[e] + beta[e]);
    const float* hrow = hprev + w * HH;
    for (int e = lane; e < HH; e += 32)
      smem[w * H2 + HH + e] = f32_to_bf16(hrow[e]);
  }
  __syncthreads();

  // ---- Phase 2: GEMM. wave = (gate g, K-quarter kq), 512 K per wave.
  int g = w & 3, kq = w >> 2;
  int j0 = blockIdx.x * 16;
  const unsigned short* arow = smem + (lane & 15) * H2 + kq * 512;
  const unsigned short* brow =
      Wm + (long long)(g * HH + j0 + (lane & 15)) * H2 + kq * 512;

  v8f acc = {0.f, 0.f, 0.f, 0.f, 0.f, 0.f, 0.f, 0.f};
#pragma unroll 4
  for (int ks = 0; ks < 16; ++ks) {
    v16bf fa = load_a_frag(arow + ks * 32, lane);
    v16bf fb = load_b_frag(brow + ks * 32, lane);
    acc = __builtin_amdgcn_wmma_f32_16x16x32_bf16(false, fa, false, fb,
                                                  (short)0, acc, false, false);
  }

  // ---- Phase 3: reduce partials across K-quarters (LDS reused; A dead now)
  __syncthreads();
  float* C = (float*)(void*)smem;        // 16 waves x 256 = 16 KB
#pragma unroll
  for (int r = 0; r < 8; ++r) C[w * 256 + r * 32 + lane] = acc[r];
  __syncthreads();
#pragma unroll
  for (int e = 0; e < 2; ++e) {
    int i = tid + e * 512;               // i in [0,1024): kq=0 partials
    C[i] = C[i] + C[i + 1024] + C[i + 2048] + C[i + 3072];
  }
  __syncthreads();

  if (tid < 256) {
    int b = tid >> 4, j = tid & 15;
    int off = ((b & 7) << 5) + ((b >> 3) << 4) + j;  // (M=r+8*(lane/16), N=lane%16)
    int col = j0 + j;
    float zi = C[off]       + bias[col];
    float zf = C[256 + off] + bias[HH + col];
    float zo = C[512 + off] + bias[2 * HH + col];
    float zg = C[768 + off] + bias[3 * HH + col];
    float si = 1.f / (1.f + __expf(-zi));
    float sf = 1.f / (1.f + __expf(-zf));
    float so = 1.f / (1.f + __expf(-zo));
    float tg = tanhf(zg);
    int idx = b * HH + col;
    float cn = sf * cstate[idx] + si * tg;
    float hn = so * tanhf(cn);
    cstate[idx] = cn;
    hout[idx] = hn;
    if (inp_out)  inp_out[idx] = hn + resid[idx];
    if (hseq_out) hseq_out[(long long)b * SS * HH + col] = f32_to_bf16(hn);
  }
}

// ---- logits: (8192 x 1024) bf16 x (1024 x 32000) bf16 -> fp32.
// Wave tile M=16,N=64 (4 accumulators, A frag reused). Grid exact: EXEC all-1s.
__global__ void k_logits(const unsigned short* __restrict__ Aseq,
                         const unsigned short* __restrict__ Eb,
                         float* __restrict__ out) {
  int tid = threadIdx.x, lane = tid & 31, w = tid >> 5;
  int gw = blockIdx.x * 8 + w;        // 0..255999
  int mt = gw / 500;                  // 0..511
  int nt = gw % 500;                  // 0..499
  const unsigned short* arow = Aseq + (long long)(mt * 16 + (lane & 15)) * HH;
  const unsigned short* br = Eb + (long long)(nt * 64 + (lane & 15)) * HH;

  v8f a0 = {0.f,0.f,0.f,0.f,0.f,0.f,0.f,0.f};
  v8f a1 = a0, a2 = a0, a3 = a0;
#pragma unroll 4
  for (int ks = 0; ks < 32; ++ks) {
    v16bf fa = load_a_frag(arow + ks * 32, lane);
    v16bf b0 = load_b_frag(br + ks * 32, lane);
    v16bf b1 = load_b_frag(br + 16ll * HH + ks * 32, lane);
    v16bf b2 = load_b_frag(br + 32ll * HH + ks * 32, lane);
    v16bf b3 = load_b_frag(br + 48ll * HH + ks * 32, lane);
    a0 = __builtin_amdgcn_wmma_f32_16x16x32_bf16(false, fa, false, b0, (short)0, a0, false, false);
    a1 = __builtin_amdgcn_wmma_f32_16x16x32_bf16(false, fa, false, b1, (short)0, a1, false, false);
    a2 = __builtin_amdgcn_wmma_f32_16x16x32_bf16(false, fa, false, b2, (short)0, a2, false, false);
    a3 = __builtin_amdgcn_wmma_f32_16x16x32_bf16(false, fa, false, b3, (short)0, a3, false, false);
  }
  int n0 = nt * 64 + (lane & 15);
#pragma unroll
  for (int r = 0; r < 8; ++r) {
    int m = mt * 16 + r + ((lane & 16) ? 8 : 0);
    float* o = out + (long long)m * VV + n0;
    o[0] = a0[r]; o[16] = a1[r]; o[32] = a2[r]; o[48] = a3[r];
  }
}

extern "C" void kernel_launch(void* const* d_in, const int* in_sizes, int n_in,
                              void* d_out, int out_size, void* d_ws, size_t ws_size,
                              hipStream_t stream) {
  const int*   x     = (const int*)d_in[0];
  const float* embed = (const float*)d_in[1];
  const float* gamma = (const float*)d_in[2];
  const float* beta  = (const float*)d_in[3];
  const float* W     = (const float*)d_in[4];
  const float* bias  = (const float*)d_in[5];
  float* out = (float*)d_out;           // [0,32768): c_final ; then logits
  char* ws = (char*)d_ws;

  unsigned short* Wt    = (unsigned short*)(ws);              //  33,554,432 B
  unsigned short* Eb    = (unsigned short*)(ws + 33554432);   //  65,536,000 B
  float*          xe    = (float*)(ws + 99090432);            //  33,554,432 B
  unsigned short* hseq  = (unsigned short*)(ws + 132644864);  //  16,777,216 B
  float*          hbuf  = (float*)(ws + 149422080);           //     262,144 B (L x 2 x B x H)
  float*          inpcur= (float*)(ws + 149684224);           //      65,536 B

  hipMemsetAsync(out, 0, (size_t)NL * BB * HH * sizeof(float), stream);       // c = 0
  hipMemsetAsync(hbuf, 0, (size_t)NL * 2 * BB * HH * sizeof(float), stream);  // h = 0

  k_cvt_w<<<(NL * H4 * H2) / 256, 256, 0, stream>>>(W, Wt);
  k_cvt_embed<<<((long long)VV * HH) / 256, 256, 0, stream>>>(embed, Eb);
  k_gather<<<(SS * BB * HH) / 256, 256, 0, stream>>>(x, embed, xe);

  for (int t = 0; t < SS; ++t) {
    const float* xet = xe + (size_t)t * BB * HH;
    int p = t & 1;
    float* h0in  = hbuf + (size_t)(0 + p)     * BB * HH;
    float* h0out = hbuf + (size_t)(0 + 1 - p) * BB * HH;
    float* h1in  = hbuf + (size_t)(2 + p)     * BB * HH;
    float* h1out = hbuf + (size_t)(2 + 1 - p) * BB * HH;
    // layer 0
    k_lstm_step<<<64, 512, 0, stream>>>(xet, gamma, beta, h0in, Wt, bias,
                                        out, h0out, xet, inpcur, nullptr);
    // layer 1
    k_lstm_step<<<64, 512, 0, stream>>>(inpcur, gamma + HH, beta + HH, h1in,
                                        Wt + (size_t)H4 * H2, bias + H4,
                                        out + BB * HH, h1out,
                                        nullptr, nullptr, hseq + (size_t)t * HH);
  }

  k_logits<<<32000, 256, 0, stream>>>(hseq, Eb, out + (size_t)NL * BB * HH);
}
